// Encoder_77017353552604
// MI455X (gfx1250) — compile-verified, matrix operand order checked by reading, not run on hardware
//
#include <hip/hip_runtime.h>
#include <hip/hip_bf16.h>

// ---------------------------------------------------------------------------
// LSTM encoder for MI455X (gfx1250, wave32, WMMA).
//   B=256, S=256, E=256, H=1024, 4H=4096, 2E=512
//   prep_kernel : fp32->bf16 conversion of weights/embeddings, bias sum, h0=0.
//   xproj_kernel: x_proj[s,b,4H] = gathered-x @ W_ih^T + bias (bf16 out).
//                 64x64 slab per wave (4x4 register blocking, 16 WMMA/k-step).
//   step_kernel : gates = xproj[t] + h @ W_hh^T with W_hh k-chunks staged in
//                 LDS (double-buffered) so W_hh is read ~2x/step instead of
//                 16x; fused sigmoid/tanh + c/h update. 256 launches.
// ---------------------------------------------------------------------------

typedef __attribute__((ext_vector_type(16))) __bf16        v16bf;
typedef __attribute__((ext_vector_type(8)))  float         v8f;
typedef __attribute__((ext_vector_type(4)))  unsigned int  u32x4;

#define B_   256
#define S_   256
#define E_   256
#define H_   1024
#define G4H  4096   // 4*H
#define KIH  512    // 2*E

// Workspace layout (bytes), all regions 256B-aligned. Total ~550.5 MB.
#define WS_WHH   0ull            // 4096*1024 bf16  =   8,388,608
#define WS_WIH   8388608ull      // 4096*512  bf16  =   4,194,304
#define WS_LEMB  12582912ull     // 30*256    bf16  =      15,360
#define WS_SEMB  12598272ull     // 4*256     bf16  =       2,048
#define WS_BIAS  12600320ull     // 4096      f32   =      16,384
#define WS_HBF   12616704ull     // 2*256*1024 bf16 =   1,048,576
#define WS_XPROJ 13665280ull     // 256*256*4096 bf16 = 536,870,912

__device__ __forceinline__ unsigned short f32_to_bf16_rne(float x) {
    unsigned u = __float_as_uint(x);
    u += 0x7FFFu + ((u >> 16) & 1u);
    return (unsigned short)(u >> 16);
}
__device__ __forceinline__ float bf16_to_f32(unsigned short h) {
    return __uint_as_float(((unsigned)h) << 16);
}
// One WMMA fragment slice per lane: two contiguous 8-elem (16B) bf16 chunks,
// per the CDNA5 ISA 16-bit A(16x32)/B(32x16) VGPR layouts.
__device__ __forceinline__ v16bf load_frag(const unsigned short* p0,
                                           const unsigned short* p1) {
    union { u32x4 u[2]; v16bf v; } r;
    r.u[0] = *(const u32x4*)p0;
    r.u[1] = *(const u32x4*)p1;
    return r.v;
}
__device__ __forceinline__ float fast_sigmoid(float x) {
    return 1.0f / (1.0f + __expf(-x));
}
__device__ __forceinline__ float fast_tanh(float x) {
    return 2.0f / (1.0f + __expf(-2.0f * x)) - 1.0f;
}

// ---------------------------------------------------------------------------
// Kernel 1: one-time data prep (bf16 conversions, bias sum, zero h0).
// ---------------------------------------------------------------------------
__global__ void prep_kernel(const float* __restrict__ letter_emb,
                            const float* __restrict__ state_emb,
                            const float* __restrict__ W_ih,
                            const float* __restrict__ W_hh,
                            const float* __restrict__ b_ih,
                            const float* __restrict__ b_hh,
                            unsigned short* __restrict__ ws_whh,
                            unsigned short* __restrict__ ws_wih,
                            unsigned short* __restrict__ ws_lemb,
                            unsigned short* __restrict__ ws_semb,
                            float* __restrict__ ws_bias,
                            unsigned short* __restrict__ ws_hbf0) {
    const unsigned NWHH = 4096u * 1024u;
    const unsigned NWIH = 4096u * 512u;
    const unsigned NLE  = 30u * 256u;
    const unsigned NSE  = 4u * 256u;
    const unsigned NB   = 4096u;
    const unsigned NH0  = (unsigned)(B_ * H_);
    const unsigned TOT  = NWHH + NWIH + NLE + NSE + NB + NH0;

    for (unsigned i = blockIdx.x * blockDim.x + threadIdx.x; i < TOT;
         i += gridDim.x * blockDim.x) {
        unsigned j = i;
        if (j < NWHH)            { ws_whh[j]  = f32_to_bf16_rne(W_hh[j]); continue; }
        j -= NWHH;
        if (j < NWIH)            { ws_wih[j]  = f32_to_bf16_rne(W_ih[j]); continue; }
        j -= NWIH;
        if (j < NLE)             { ws_lemb[j] = f32_to_bf16_rne(letter_emb[j]); continue; }
        j -= NLE;
        if (j < NSE)             { ws_semb[j] = f32_to_bf16_rne(state_emb[j]); continue; }
        j -= NSE;
        if (j < NB)              { ws_bias[j] = b_ih[j] + b_hh[j]; continue; }
        j -= NB;
        ws_hbf0[j] = 0;  // h0 = 0 (bf16 zero)
    }
}

// ---------------------------------------------------------------------------
// Kernel 2: x_proj[s*B+b, 4H] = gathered-x @ W_ih^T + bias, bf16 output.
// M=65536, N=4096, K=512. Each wave: 64x64 slab (4 m-tiles x 4 n-tiles,
// 16 accumulators, 16 WMMA per 32-wide k-step at ~1 load per WMMA).
// A rows are embedding rows (34 distinct) -> pure cache hits; 4x m-blocking
// cuts W_ih L2 traffic 4x vs the naive 1x4 mapping.
// 65536 waves = 8192 blocks x 8 waves.
// ---------------------------------------------------------------------------
__global__ void __launch_bounds__(256)
xproj_kernel(const int* __restrict__ letter_seq, const int* __restrict__ state_seq,
             const unsigned short* __restrict__ lemb,
             const unsigned short* __restrict__ semb,
             const unsigned short* __restrict__ wih,
             const float* __restrict__ bias,
             unsigned short* __restrict__ xproj) {
    const int lane = threadIdx.x & 31;
    const int wid  = (blockIdx.x * blockDim.x + threadIdx.x) >> 5; // 0..65535
    const int mg   = wid >> 6;   // 0..1023 : 64-row group of M
    const int ng   = wid & 63;   // 0..63   : 64-col group of N
    const int l16  = lane & 15;
    const int hi   = lane >> 4;
    const int kb   = hi * 8;
    const int n0   = ng * 64;

    // Per-lane embedding row pointers for the 4 m-tiles.
    const unsigned short* lrow[4];
    const unsigned short* srow[4];
#pragma unroll
    for (int mi = 0; mi < 4; ++mi) {
        const int m = mg * 64 + mi * 16 + l16;
        const int b = m >> 8, s = m & 255;
        lrow[mi] = lemb + (size_t)letter_seq[b * S_ + s] * E_;
        srow[mi] = semb + (size_t)state_seq[b * S_ + s] * E_;
    }
    const unsigned short* wr[4];
#pragma unroll
    for (int ni = 0; ni < 4; ++ni)
        wr[ni] = wih + (size_t)(n0 + ni * 16 + l16) * KIH;

    v8f acc[4][4];
#pragma unroll
    for (int mi = 0; mi < 4; ++mi)
#pragma unroll
        for (int ni = 0; ni < 4; ++ni) acc[mi][ni] = (v8f){};

    // First half of K: letter embedding columns (k in [0,256)).
    for (int k0 = 0; k0 < E_; k0 += 32) {
        v16bf bf[4];
#pragma unroll
        for (int ni = 0; ni < 4; ++ni)
            bf[ni] = load_frag(wr[ni] + k0 + kb, wr[ni] + k0 + 16 + kb);
#pragma unroll
        for (int mi = 0; mi < 4; ++mi) {
            v16bf a = load_frag(lrow[mi] + k0 + kb, lrow[mi] + k0 + 16 + kb);
#pragma unroll
            for (int ni = 0; ni < 4; ++ni)
                acc[mi][ni] = __builtin_amdgcn_wmma_f32_16x16x32_bf16(
                    false, a, false, bf[ni], (short)0, acc[mi][ni], false, false);
        }
    }
    // Second half of K: state embedding columns (k in [256,512)).
    for (int k0 = E_; k0 < KIH; k0 += 32) {
        const int ks = k0 - E_;
        v16bf bf[4];
#pragma unroll
        for (int ni = 0; ni < 4; ++ni)
            bf[ni] = load_frag(wr[ni] + k0 + kb, wr[ni] + k0 + 16 + kb);
#pragma unroll
        for (int mi = 0; mi < 4; ++mi) {
            v16bf a = load_frag(srow[mi] + ks + kb, srow[mi] + ks + 16 + kb);
#pragma unroll
            for (int ni = 0; ni < 4; ++ni)
                acc[mi][ni] = __builtin_amdgcn_wmma_f32_16x16x32_bf16(
                    false, a, false, bf[ni], (short)0, acc[mi][ni], false, false);
        }
    }

    // Store: D layout -> VGPR j, lane l : row = 16*tile + j + 8*(l/16), col = l%16.
    float bi[4];
#pragma unroll
    for (int ni = 0; ni < 4; ++ni) bi[ni] = bias[n0 + ni * 16 + l16];
#pragma unroll
    for (int mi = 0; mi < 4; ++mi) {
#pragma unroll
        for (int j = 0; j < 8; ++j) {
            const int mr = mg * 64 + mi * 16 + j + 8 * hi;
            const size_t base =
                ((size_t)((mr & 255) * B_ + (mr >> 8))) * G4H;  // (s*B + b)*4H
#pragma unroll
            for (int ni = 0; ni < 4; ++ni)
                xproj[base + n0 + ni * 16 + l16] =
                    f32_to_bf16_rne(acc[mi][ni][j] + bi[ni]);
        }
    }
}

// ---------------------------------------------------------------------------
// Kernel 3: one LSTM timestep, fused GEMM + nonlinearity + state update.
// Grid: 128 blocks = 64 nt-tiles x 2 batch-halves; 8 waves/block, wave w owns
// batch tile mt = mhalf*8 + w. The four W_hh gate k-chunks for this nt are
// cooperatively staged in LDS (4KB/k-step, double buffered) and shared by all
// 8 waves -> W_hh global traffic is 2x8.4MB per step instead of 16x.
// ---------------------------------------------------------------------------
__global__ void __launch_bounds__(256)
lstm_step_kernel(const unsigned short* __restrict__ xproj,
                 const unsigned short* __restrict__ whh,
                 const unsigned short* __restrict__ hprev_bf,
                 unsigned short* __restrict__ hnext_bf,
                 float* __restrict__ hidden_out,
                 float* __restrict__ cell_out,
                 int t) {
    __shared__ unsigned short ldsB[2][4 * 16 * 32];  // [buf][gate*16+row][k 0..31]

    const int tid  = threadIdx.x;
    const int lane = tid & 31;
    const int nt    = blockIdx.x & 63;   // hidden 16-col tile
    const int mhalf = blockIdx.x >> 6;   // batch half
    const int mt    = mhalf * 8 + (tid >> 5);  // this wave's batch 16-row tile
    const int l16  = lane & 15;
    const int hi   = lane >> 4;
    const int kb   = hi * 8;

    // Cooperative W_hh->LDS staging: 256 threads x 16B = 4KB per k-step.
    const int cg = tid >> 6;          // gate 0..3
    const int cr = (tid >> 2) & 15;   // row in tile 0..15
    const int ck = (tid & 3) * 8;     // k-offset 0,8,16,24
    const unsigned short* gsrc =
        whh + (size_t)(cg * H_ + nt * 16 + cr) * H_ + ck;
    unsigned short* ldst0 = &ldsB[0][(cg * 16 + cr) * 32 + ck];
    unsigned short* ldst1 = &ldsB[1][(cg * 16 + cr) * 32 + ck];

    // Seed accumulators with x_proj[t] (already includes b_ih + b_hh).
    v8f acc[4];
#pragma unroll
    for (int g = 0; g < 4; ++g) {
        const int n = g * H_ + nt * 16 + l16;
#pragma unroll
        for (int j = 0; j < 8; ++j) {
            const int m = mt * 16 + j + 8 * hi;
            acc[g][j] = bf16_to_f32(xproj[((size_t)(t * B_ + m)) * G4H + n]);
        }
    }

    const unsigned short* arow = hprev_bf + (size_t)(mt * 16 + l16) * H_;

    // Prologue: stage k-chunk 0.
    *(u32x4*)ldst0 = *(const u32x4*)gsrc;
    __syncthreads();

    for (int it = 0; it < 32; ++it) {
        const int k0 = it * 32;
        // Prefetch next W_hh chunk to registers (overlaps with WMMA below).
        u32x4 nxt;
        if (it + 1 < 32) nxt = *(const u32x4*)(gsrc + k0 + 32);

        const unsigned short* lb = &ldsB[it & 1][0];
        v16bf a  = load_frag(arow + k0 + kb, arow + k0 + 16 + kb);
        v16bf w0 = load_frag(lb + (0 * 16 + l16) * 32 + kb, lb + (0 * 16 + l16) * 32 + 16 + kb);
        v16bf w1 = load_frag(lb + (1 * 16 + l16) * 32 + kb, lb + (1 * 16 + l16) * 32 + 16 + kb);
        v16bf w2 = load_frag(lb + (2 * 16 + l16) * 32 + kb, lb + (2 * 16 + l16) * 32 + 16 + kb);
        v16bf w3 = load_frag(lb + (3 * 16 + l16) * 32 + kb, lb + (3 * 16 + l16) * 32 + 16 + kb);
        acc[0] = __builtin_amdgcn_wmma_f32_16x16x32_bf16(false, a, false, w0, (short)0, acc[0], false, false);
        acc[1] = __builtin_amdgcn_wmma_f32_16x16x32_bf16(false, a, false, w1, (short)0, acc[1], false, false);
        acc[2] = __builtin_amdgcn_wmma_f32_16x16x32_bf16(false, a, false, w2, (short)0, acc[2], false, false);
        acc[3] = __builtin_amdgcn_wmma_f32_16x16x32_bf16(false, a, false, w3, (short)0, acc[3], false, false);

        // Store prefetched chunk into the *other* buffer, then one barrier:
        // compute(cur) -> store(next) -> barrier makes every cross-iteration
        // read/write pair on a buffer straddle exactly one barrier.
        if (it + 1 < 32) *(u32x4*)((it & 1) ? ldst0 : ldst1) = nxt;
        __syncthreads();
    }

    // Fused elementwise: i,f,o = sigmoid; g = tanh; c = f*c + i*g; h = o*tanh(c).
#pragma unroll
    for (int j = 0; j < 8; ++j) {
        const int m = mt * 16 + j + 8 * hi;
        const int n = nt * 16 + l16;
        const float iv = fast_sigmoid(acc[0][j]);
        const float fv = fast_sigmoid(acc[1][j]);
        const float gv = fast_tanh(acc[2][j]);
        const float ov = fast_sigmoid(acc[3][j]);
        const size_t oidx = (size_t)m * (S_ * H_) + (size_t)t * H_ + n;
        const float cprev = (t == 0) ? 0.0f : cell_out[oidx - H_];
        const float c = fv * cprev + iv * gv;
        const float h = ov * fast_tanh(c);
        hidden_out[oidx] = h;
        cell_out[oidx]   = c;
        hnext_bf[m * H_ + n] = f32_to_bf16_rne(h);
    }
}

// ---------------------------------------------------------------------------
extern "C" void kernel_launch(void* const* d_in, const int* in_sizes, int n_in,
                              void* d_out, int out_size, void* d_ws, size_t ws_size,
                              hipStream_t stream) {
    const int*   letter_seq = (const int*)d_in[0];
    const int*   state_seq  = (const int*)d_in[1];
    const float* letter_emb = (const float*)d_in[2];
    const float* state_emb  = (const float*)d_in[3];
    const float* W_ih       = (const float*)d_in[4];
    const float* W_hh       = (const float*)d_in[5];
    const float* b_ih       = (const float*)d_in[6];
    const float* b_hh       = (const float*)d_in[7];

    char* ws = (char*)d_ws;
    unsigned short* ws_whh   = (unsigned short*)(ws + WS_WHH);
    unsigned short* ws_wih   = (unsigned short*)(ws + WS_WIH);
    unsigned short* ws_lemb  = (unsigned short*)(ws + WS_LEMB);
    unsigned short* ws_semb  = (unsigned short*)(ws + WS_SEMB);
    float*          ws_bias  = (float*)(ws + WS_BIAS);
    unsigned short* ws_hbf   = (unsigned short*)(ws + WS_HBF);
    unsigned short* ws_xproj = (unsigned short*)(ws + WS_XPROJ);

    float* hidden_out = (float*)d_out;
    float* cell_out   = hidden_out + (size_t)B_ * S_ * H_;

    prep_kernel<<<2048, 256, 0, stream>>>(letter_emb, state_emb, W_ih, W_hh,
                                          b_ih, b_hh, ws_whh, ws_wih, ws_lemb,
                                          ws_semb, ws_bias, ws_hbf);

    // 65536 waves: 8192 blocks x 8 waves.
    xproj_kernel<<<8192, 256, 0, stream>>>(letter_seq, state_seq, ws_lemb,
                                           ws_semb, ws_wih, ws_bias, ws_xproj);

    // Sequential scan: h ping-pongs between two bf16 buffers; c is read back
    // from the cell output written at t-1 (deterministic within one call).
    for (int t = 0; t < S_; ++t) {
        const unsigned short* hprev = ws_hbf + (size_t)(t & 1) * (B_ * H_);
        unsigned short*       hnext = ws_hbf + (size_t)((t + 1) & 1) * (B_ * H_);
        lstm_step_kernel<<<128, 256, 0, stream>>>(ws_xproj, ws_whh, hprev, hnext,
                                                  hidden_out, cell_out, t);
    }
}
